// SwinTransformerBlock_11888469475359
// MI455X (gfx1250) — matrix operand order repaired and checked by reading, not compiled, not run on hardware
//
#include <hip/hip_runtime.h>
#include <hip/hip_bf16.h>
#include <math.h>

typedef _Float16 f16;
typedef __attribute__((ext_vector_type(16))) _Float16 v16h;
typedef __attribute__((ext_vector_type(8)))  _Float16 v8h;
typedef __attribute__((ext_vector_type(4)))  _Float16 v4h;
typedef __attribute__((ext_vector_type(8)))  float    v8f;
typedef __attribute__((ext_vector_type(4)))  int      v4i;

// ---------------- problem constants ----------------
#define BDIM   32
#define HDIM   56
#define CDIM   192
#define WS     7
#define SHIFT  3
#define NH     6
#define HD     32
#define NTOK   49                 // tokens per window
#define NWIN   64                 // windows per image (8x8)
#define MTOT   (BDIM * NWIN * NTOK)   // 100352 rows, multiple of 128

// ---------------- workspace layout (bytes) ----------------
#define OFF_WQKV  ((size_t)0)                         // 576*192*2   = 221184
#define OFF_WOUT  ((size_t)221184)                    // 192*192*2   =  73728
#define OFF_WFC1  ((size_t)294912)                    // 768*192*2   = 294912
#define OFF_WFC2  ((size_t)589824)                    // 192*768*2   = 294912
#define OFF_XW    ((size_t)884736)                    // 100352*192*2 = 38535168  (reused as attn_out)
#define OFF_QKVH  ((size_t)39419904)                  // 2048*6*3*49*32*2 = 115605504 (reused as h2)
#define OFF_X1    ((size_t)155025408)                 // 100352*192*4 = 77070336
#define OFF_Y1    ((size_t)232095744)                 // 100352*768*2 = 154140672

// ---------------- CDNA5 async global->LDS staging (guarded) ----------------
#if defined(__gfx1250__) && __has_builtin(__builtin_amdgcn_global_load_async_to_lds_b128)
#define HAVE_ASYNC_LDS 1
#define AS1 __attribute__((address_space(1)))
#define AS3 __attribute__((address_space(3)))
#else
#define HAVE_ASYNC_LDS 0
#endif

__device__ __forceinline__ void cp_b128_g2l(f16* lds_dst, const f16* gsrc) {
#if HAVE_ASYNC_LDS
  __builtin_amdgcn_global_load_async_to_lds_b128(
      (AS1 v4i*)(AS1 void*)gsrc, (AS3 v4i*)(AS3 void*)lds_dst, 0, 0);
#else
  *(v8h*)lds_dst = *(const v8h*)gsrc;
#endif
}

__device__ __forceinline__ void wait_async_lds() {
#if HAVE_ASYNC_LDS
#if __has_builtin(__builtin_amdgcn_s_wait_asynccnt)
  __builtin_amdgcn_s_wait_asynccnt(0);
#else
  asm volatile("s_wait_asynccnt 0x0" ::: "memory");
#endif
#endif
}

// ---------------- WMMA fragment loaders (CDNA5 layouts) ----------------
// A 16x32 f16: lane r=l%16 = row, h=l/16; elems j<8 -> K=8h+j ; j>=8 -> K=16+8h+(j-8)
__device__ __forceinline__ v16h ld_a_frag(const f16* base, int ld) {
  int lane = threadIdx.x & 31;
  int r = lane & 15, h = lane >> 4;
  const f16* p = base + r * ld + 8 * h;
  v8h lo = *(const v8h*)(p);
  v8h hi = *(const v8h*)(p + 16);
  v16h a;
#pragma unroll
  for (int j = 0; j < 8; ++j) { a[j] = lo[j]; a[j + 8] = hi[j]; }
  return a;
}
// B 32x16 f16 from a row-major [N][K] tile (B[k][n] = tile[n][k]):
// lane c=l%16 = column, h=l/16; elems j<8 -> K=8h+j ; j>=8 -> K=16+8h+(j-8)
__device__ __forceinline__ v16h ld_b_frag(const f16* base, int ld) {
  int lane = threadIdx.x & 31;
  int c = lane & 15, h = lane >> 4;
  const f16* p = base + c * ld + 8 * h;
  v8h lo = *(const v8h*)(p);
  v8h hi = *(const v8h*)(p + 16);
  v16h b;
#pragma unroll
  for (int j = 0; j < 8; ++j) { b[j] = lo[j]; b[j + 8] = hi[j]; }
  return b;
}

__device__ __forceinline__ v8f wmma_f16(v16h a, v16h b, v8f c) {
  return __builtin_amdgcn_wmma_f32_16x16x32_f16(false, a, false, b, (short)0, c,
                                                false, false);
}

// ---------------- f32 -> f16 conversion ----------------
__global__ __launch_bounds__(256) void cvt_f16_kernel(const float* __restrict__ src,
                                                      f16* __restrict__ dst, int n) {
  int i = blockIdx.x * 256 + threadIdx.x;
  if (i < n) dst[i] = (f16)src[i];
}

// ---------------- LN1 + shift-roll + window partition ----------------
// row m = win*49 + n ; win = b*64 + w ; source (h,w) = shifted coords + SHIFT mod 56
__global__ __launch_bounds__(256) void ln_shift_kernel(const float* __restrict__ x,
                                                       const float* __restrict__ g,
                                                       const float* __restrict__ beta,
                                                       f16* __restrict__ out) {
  int wid = threadIdx.x >> 5, lane = threadIdx.x & 31;
  int m = blockIdx.x * 8 + wid;
  if (m >= MTOT) return;
  int win = m / NTOK, n = m % NTOK;
  int b_ = win >> 6, w = win & 63;
  int wr = w >> 3, wc = w & 7, tr = n / 7, tc = n % 7;
  int hh = (wr * 7 + tr + SHIFT) % HDIM;
  int ww = (wc * 7 + tc + SHIFT) % HDIM;
  const float* src = x + (((size_t)b_ * HDIM + hh) * HDIM + ww) * CDIM;
  float vals[6], s = 0.f, s2 = 0.f;
#pragma unroll
  for (int i = 0; i < 6; ++i) {
    float v = src[lane + 32 * i];
    vals[i] = v; s += v; s2 += v * v;
  }
#pragma unroll
  for (int o = 16; o; o >>= 1) { s += __shfl_xor(s, o); s2 += __shfl_xor(s2, o); }
  float mu = s * (1.f / CDIM);
  float var = s2 * (1.f / CDIM) - mu * mu;
  float rstd = rsqrtf(var + 1e-5f);
#pragma unroll
  for (int i = 0; i < 6; ++i) {
    int c = lane + 32 * i;
    out[(size_t)m * CDIM + c] = (f16)((vals[i] - mu) * rstd * g[c] + beta[c]);
  }
}

// ---------------- LN2 (linear rows) ----------------
__global__ __launch_bounds__(256) void ln_kernel(const float* __restrict__ x,
                                                 const float* __restrict__ g,
                                                 const float* __restrict__ beta,
                                                 f16* __restrict__ out) {
  int wid = threadIdx.x >> 5, lane = threadIdx.x & 31;
  int m = blockIdx.x * 8 + wid;
  if (m >= MTOT) return;
  const float* src = x + (size_t)m * CDIM;
  float vals[6], s = 0.f, s2 = 0.f;
#pragma unroll
  for (int i = 0; i < 6; ++i) {
    float v = src[lane + 32 * i];
    vals[i] = v; s += v; s2 += v * v;
  }
#pragma unroll
  for (int o = 16; o; o >>= 1) { s += __shfl_xor(s, o); s2 += __shfl_xor(s2, o); }
  float mu = s * (1.f / CDIM);
  float var = s2 * (1.f / CDIM) - mu * mu;
  float rstd = rsqrtf(var + 1e-5f);
#pragma unroll
  for (int i = 0; i < 6; ++i) {
    int c = lane + 32 * i;
    out[(size_t)m * CDIM + c] = (f16)((vals[i] - mu) * rstd * g[c] + beta[c]);
  }
}

// ---------------- generic WMMA GEMM: D = A[M][K] * W[N][K]^T + bias ----------------
// Block tile 128(M) x 64(N); 8 waves as 4x2; each wave owns a 2x2 grid of 16x16
// tiles (4 WMMA per K-step with a0/a1 x b0/b1 fragment reuse). K-chunks of 32
// double-buffered through LDS with async global->LDS staging (one barrier/step).
// Epilogues: 1 = q/k/v scatter (q scaled), 2 = out-proj + unshift + residual -> f32,
//            3 = exact GELU -> f16, 4 = +resid(linear) -> f32.
template <int MODE, int K>
__global__ __launch_bounds__(256) void gemm_wmma_kernel(
    const f16* __restrict__ A, const f16* __restrict__ W,
    const float* __restrict__ bias,
    f16* __restrict__ out16, float* __restrict__ out32,
    const float* __restrict__ resid, int ldo) {
  constexpr int NK = K / 32;
  __shared__ f16 sA[2][128][32];
  __shared__ f16 sB[2][64][32];
  int m0 = blockIdx.x * 128;
  int n0 = blockIdx.y * 64;
  int tid = threadIdx.x;
  int wid = tid >> 5, lane = tid & 31;
  int wm = wid >> 1, wn = wid & 1;

  // staging coordinates: A = 128x32 halfs (2 x b128/thread), B = 64x32 (1 x b128)
  int ra = tid >> 1, koa = (tid & 1) * 16;
  int rb = tid >> 2, kob = (tid & 3) * 8;
  const f16* gA = A + (size_t)(m0 + ra) * K + koa;
  const f16* gB = W + (size_t)(n0 + rb) * K + kob;

  v8f acc[2][2] = {};

  // preload k-chunk 0 into buffer 0
  cp_b128_g2l(&sA[0][ra][koa], gA);
  cp_b128_g2l(&sA[0][ra][koa + 8], gA + 8);
  cp_b128_g2l(&sB[0][rb][kob], gB);
  wait_async_lds();
  __syncthreads();

#pragma unroll
  for (int ki = 0; ki < NK; ++ki) {
    int cur = ki & 1;
    if (ki + 1 < NK) {  // stage next chunk into the other buffer (async)
      int k0 = (ki + 1) * 32;
      cp_b128_g2l(&sA[cur ^ 1][ra][koa], gA + k0);
      cp_b128_g2l(&sA[cur ^ 1][ra][koa + 8], gA + k0 + 8);
      cp_b128_g2l(&sB[cur ^ 1][rb][kob], gB + k0);
    }
    v16h a0 = ld_a_frag(&sA[cur][wm * 32][0], 32);
    v16h a1 = ld_a_frag(&sA[cur][wm * 32 + 16][0], 32);
    v16h b0 = ld_b_frag(&sB[cur][wn * 32][0], 32);
    v16h b1 = ld_b_frag(&sB[cur][wn * 32 + 16][0], 32);
    acc[0][0] = wmma_f16(a0, b0, acc[0][0]);
    acc[0][1] = wmma_f16(a0, b1, acc[0][1]);
    acc[1][0] = wmma_f16(a1, b0, acc[1][0]);
    acc[1][1] = wmma_f16(a1, b1, acc[1][1]);
    wait_async_lds();
    __syncthreads();
  }

  int c = lane & 15, h = lane >> 4;
#pragma unroll
  for (int tm = 0; tm < 2; ++tm) {
#pragma unroll
    for (int tn = 0; tn < 2; ++tn) {
      int ncol = n0 + wn * 32 + tn * 16 + c;
      float bv = bias[ncol];
#pragma unroll
      for (int i = 0; i < 8; ++i) {
        int m = m0 + wm * 32 + tm * 16 + 8 * h + i;
        float val = acc[tm][tn][i] + bv;
        if (MODE == 1) {               // qkv scatter: [win][head][which][n][hd]
          int win = m / NTOK, n = m % NTOK;
          int which = ncol / CDIM, cc = ncol % CDIM;
          int head = cc >> 5, hd = cc & 31;
          if (which == 0) val *= 0.17677669529663687f;  // HD^-0.5
          out16[((((size_t)win * NH + head) * 3 + which) * NTOK + n) * HD + hd] = (f16)val;
        } else if (MODE == 2) {        // out-proj: merge + reverse roll + residual
          int win = m / NTOK, n = m % NTOK;
          int b_ = win >> 6, w = win & 63;
          int wr = w >> 3, wc = w & 7, tr = n / 7, tc = n % 7;
          int hh = (wr * 7 + tr + SHIFT) % HDIM;
          int ww = (wc * 7 + tc + SHIFT) % HDIM;
          size_t di = (((size_t)b_ * HDIM + hh) * HDIM + ww) * CDIM + ncol;
          out32[di] = resid[di] + val;
        } else if (MODE == 3) {        // exact gelu
          float gv = 0.5f * val * (1.0f + erff(val * 0.70710678118654752f));
          out16[(size_t)m * ldo + ncol] = (f16)gv;
        } else {                       // MODE 4: fc2 + residual
          size_t di = (size_t)m * ldo + ncol;
          out32[di] = resid[di] + val;
        }
      }
    }
  }
}

// ---------------- windowed attention, one block per (window, head) ----------------
__global__ __launch_bounds__(128) void attn_kernel(const f16* __restrict__ qkvh,
                                                   const float* __restrict__ rpb,
                                                   f16* __restrict__ attn_out) {
  __shared__ f16 sQ[64][32];
  __shared__ f16 sK[64][32];
  __shared__ f16 sVT[32][64];
  __shared__ float sS[64][64];
  __shared__ f16 sP[64][64];

  int bid = blockIdx.x;           // win*NH + head
  int win = bid / NH, head = bid % NH;
  int tid = threadIdx.x;
  const f16* qg = qkvh + (size_t)bid * 3 * NTOK * HD;
  const f16* kg = qg + NTOK * HD;
  const f16* vg = kg + NTOK * HD;

  // q/k rows: vectorized 16B chunks, zero-pad rows >= 49
#pragma unroll
  for (int t = 0; t < 2; ++t) {
    int chunk = tid + t * 128;           // 256 chunks of 8 halfs
    int n = chunk >> 2, ko = (chunk & 3) * 8;
    v8h z = {};
    *(v8h*)&sQ[n][ko] = (n < NTOK) ? *(const v8h*)(qg + n * HD + ko) : z;
    *(v8h*)&sK[n][ko] = (n < NTOK) ? *(const v8h*)(kg + n * HD + ko) : z;
  }
  // v transposed (scalar scatter), zero-pad cols >= 49
  for (int idx = tid; idx < 32 * 64; idx += 128) {
    int hd = idx >> 6, n = idx & 63;
    sVT[hd][n] = (n < NTOK) ? vg[n * HD + hd] : (f16)0.f;
  }
  for (int idx = tid; idx < 64 * 64; idx += 128) ((f16*)sP)[idx] = (f16)0.f;
  __syncthreads();

  int wid = tid >> 5, lane = tid & 31;
  int c = lane & 15, h = lane >> 4;

  // S = Q @ K^T (wave wid owns row-tile wid; all 4 col tiles; K = 32)
  {
    v16h a = ld_a_frag(&sQ[wid * 16][0], 32);
#pragma unroll
    for (int ct = 0; ct < 4; ++ct) {
      v16h b = ld_b_frag(&sK[ct * 16][0], 32);
      v8f d = {};
      d = wmma_f16(a, b, d);
#pragma unroll
      for (int i = 0; i < 8; ++i) sS[wid * 16 + 8 * h + i][ct * 16 + c] = d[i];
    }
  }
  __syncthreads();

  // softmax with rel-pos bias + shift mask (one thread per query row)
  if (tid < NTOK) {
    int r = tid;
    int w = win & 63;
    int wr = w >> 3, wc = w & 7;
    int tr_r = r / 7, tc_r = r % 7;
    int hs_r = wr * 7 + tr_r, ws_r = wc * 7 + tc_r;
    int reg_r = (1 + (hs_r >= 49) + (hs_r >= 53)) + 3 * (1 + (ws_r >= 49) + (ws_r >= 53));
    float mx = -1e30f;
    for (int j = 0; j < NTOK; ++j) {
      int tr_j = j / 7, tc_j = j % 7;
      int hs_j = wr * 7 + tr_j, ws_j = wc * 7 + tc_j;
      int reg_j = (1 + (hs_j >= 49) + (hs_j >= 53)) + 3 * (1 + (ws_j >= 49) + (ws_j >= 53));
      int ridx = (tr_r - tr_j + 6) * 13 + (tc_r - tc_j + 6);
      float v = sS[r][j] + rpb[ridx * NH + head] + ((reg_j != reg_r) ? -100.f : 0.f);
      sS[r][j] = v;
      mx = fmaxf(mx, v);
    }
    float sum = 0.f;
    for (int j = 0; j < NTOK; ++j) {
      float e = __expf(sS[r][j] - mx);
      sS[r][j] = e;
      sum += e;
    }
    float inv = 1.f / sum;
    for (int j = 0; j < NTOK; ++j) sP[r][j] = (f16)(sS[r][j] * inv);
  }
  __syncthreads();

  // O = P @ V : 4x2 tiles of 16x16, K = 64 in two chunks; wave wid owns row-tile wid
#pragma unroll
  for (int nt = 0; nt < 2; ++nt) {
    v8f d = {};
#pragma unroll
    for (int kc = 0; kc < 2; ++kc) {
      v16h a = ld_a_frag(&sP[wid * 16][kc * 32], 64);
      v16h b = ld_b_frag(&sVT[nt * 16][kc * 32], 64);
      d = wmma_f16(a, b, d);
    }
#pragma unroll
    for (int i = 0; i < 8; ++i) {
      int n = wid * 16 + 8 * h + i;
      if (n < NTOK)
        attn_out[((size_t)win * NTOK + n) * CDIM + head * HD + nt * 16 + c] = (f16)d[i];
    }
  }
}

// ---------------- launcher ----------------
extern "C" void kernel_launch(void* const* d_in, const int* in_sizes, int n_in,
                              void* d_out, int out_size, void* d_ws, size_t ws_size,
                              hipStream_t stream) {
  const float* x       = (const float*)d_in[0];
  const float* w_qkv   = (const float*)d_in[1];
  const float* b_qkv   = (const float*)d_in[2];
  const float* w_out   = (const float*)d_in[3];
  const float* b_out   = (const float*)d_in[4];
  const float* rpb     = (const float*)d_in[5];
  const float* g1      = (const float*)d_in[6];
  const float* beta1   = (const float*)d_in[7];
  const float* g2      = (const float*)d_in[8];
  const float* beta2   = (const float*)d_in[9];
  const float* w_fc1   = (const float*)d_in[10];
  const float* b_fc1   = (const float*)d_in[11];
  const float* w_fc2   = (const float*)d_in[12];
  const float* b_fc2   = (const float*)d_in[13];

  char* ws = (char*)d_ws;
  f16* wqkv16 = (f16*)(ws + OFF_WQKV);
  f16* wout16 = (f16*)(ws + OFF_WOUT);
  f16* wfc116 = (f16*)(ws + OFF_WFC1);
  f16* wfc216 = (f16*)(ws + OFF_WFC2);
  f16* xw     = (f16*)(ws + OFF_XW);
  f16* attn_o = (f16*)(ws + OFF_XW);    // alias: xw dead after QKV GEMM
  f16* qkvh   = (f16*)(ws + OFF_QKVH);
  f16* h2     = (f16*)(ws + OFF_QKVH);  // alias: qkvh dead after attention
  float* x1   = (float*)(ws + OFF_X1);
  f16* y1     = (f16*)(ws + OFF_Y1);

  // 0) weights -> f16
  cvt_f16_kernel<<<(576 * 192 + 255) / 256, 256, 0, stream>>>(w_qkv, wqkv16, 576 * 192);
  cvt_f16_kernel<<<(192 * 192 + 255) / 256, 256, 0, stream>>>(w_out, wout16, 192 * 192);
  cvt_f16_kernel<<<(768 * 192 + 255) / 256, 256, 0, stream>>>(w_fc1, wfc116, 768 * 192);
  cvt_f16_kernel<<<(192 * 768 + 255) / 256, 256, 0, stream>>>(w_fc2, wfc216, 192 * 768);

  // 1) LN1 + shift + window partition
  ln_shift_kernel<<<MTOT / 8, 256, 0, stream>>>(x, g1, beta1, xw);

  // 2) QKV GEMM: M=100352, K=192, N=576 -> per-head q/k/v (q scaled)
  gemm_wmma_kernel<1, 192><<<dim3(MTOT / 128, 576 / 64), 256, 0, stream>>>(
      xw, wqkv16, b_qkv, qkvh, nullptr, nullptr, 0);

  // 3) attention per (window, head)
  attn_kernel<<<BDIM * NWIN * NH, 128, 0, stream>>>(qkvh, rpb, attn_o);

  // 4) out-proj + merge + reverse-roll + residual -> x1 (f32)
  gemm_wmma_kernel<2, 192><<<dim3(MTOT / 128, 192 / 64), 256, 0, stream>>>(
      attn_o, wout16, b_out, nullptr, x1, x, 0);

  // 5) LN2
  ln_kernel<<<MTOT / 8, 256, 0, stream>>>(x1, g2, beta2, h2);

  // 6) fc1 + exact GELU: M=100352, K=192, N=768
  gemm_wmma_kernel<3, 192><<<dim3(MTOT / 128, 768 / 64), 256, 0, stream>>>(
      h2, wfc116, b_fc1, y1, nullptr, nullptr, 768);

  // 7) fc2 + residual -> d_out: M=100352, K=768, N=192
  gemm_wmma_kernel<4, 768><<<dim3(MTOT / 128, 192 / 64), 256, 0, stream>>>(
      y1, wfc216, b_fc2, nullptr, (float*)d_out, x1, 192);

  (void)in_sizes; (void)n_in; (void)out_size; (void)ws_size;
}
